// Bi_LSTM_86809878986959
// MI455X (gfx1250) — matrix operand order repaired and checked by reading, not compiled, not run on hardware
//
#include <hip/hip_runtime.h>
#include <hip/hip_bf16.h>
#include <stdint.h>

// ---------------------------------------------------------------------------
// BiLSTM + CRF Viterbi for MI455X (gfx1250, wave32, WMMA + TDM)
// ---------------------------------------------------------------------------
#define S_LEN   4096
#define DIM     1024
#define H2      512
#define G4      2048          // 4*H2
#define T_TAGS  12
#define START_T 10
#define STOP_T  11
#define NEGV    (-10000.0f)
#define NWG_DIR 16            // persistent workgroups per direction
#define KCHUNKS (DIM / 32)    // 32 K-chunks of 32

typedef __attribute__((ext_vector_type(16))) __bf16 v16bf;
typedef __attribute__((ext_vector_type(8)))  __bf16 v8bf;
typedef __attribute__((ext_vector_type(8)))  float  v8f;
typedef unsigned int v4u __attribute__((ext_vector_type(4)));
typedef int          v8i __attribute__((ext_vector_type(8)));
typedef int          v4i __attribute__((ext_vector_type(4)));

// ---------------------------------------------------------------------------
// init: zero barrier counters, seed h double-buffers with h0
// ---------------------------------------------------------------------------
__global__ __launch_bounds__(1024)
void init_state_kernel(const float* __restrict__ h0, float* __restrict__ hbuf,
                       unsigned* __restrict__ bars) {
    int t = threadIdx.x;
    if (t < 1024) {
        int dir = t >> 9, j = t & 511;
        hbuf[dir * 1024 + 0 * 512 + j] = h0[dir * 512 + j];   // buffer 0 = h0
    }
    if (t < 2) bars[t] = 0u;
}

// ---------------------------------------------------------------------------
// embedding gather + f32 -> bf16
// ---------------------------------------------------------------------------
__global__ __launch_bounds__(1024)
void gather_embed_kernel(const int* __restrict__ sent,
                         const float* __restrict__ embed,
                         __bf16* __restrict__ xb) {
    int s = blockIdx.x;
    int d = threadIdx.x;
    int row = sent[s];
    xb[(size_t)s * DIM + d] = (__bf16)embed[(size_t)row * DIM + d];
}

__global__ __launch_bounds__(256)
void cvt_bf16_kernel(const float* __restrict__ src, __bf16* __restrict__ dst, int n) {
    int i = blockIdx.x * 256 + threadIdx.x;
    if (i < n) dst[i] = (__bf16)src[i];
}

// ---------------------------------------------------------------------------
// TDM: DMA one 64(N-rows) x 32(K, contiguous) bf16 panel of W into LDS.
// D# per CDNA5 ISA ch.8: group0 = {count, lds_addr, global_addr, type=2},
// group1 = {data_size=2B, tensor dims (huge: no OOB clip), tile 32x64,
//           tensor_dim0_stride = 1024 elements}.
// clang-23 toolchain: 6-arg builtin (g0 v4u, g1 v8i, v4i, v4i, v8i, cpol).
// ---------------------------------------------------------------------------
__device__ __forceinline__ void tdm_load_panel(const __bf16* gsrc, unsigned lds_off) {
    unsigned long long ga = (unsigned long long)(uintptr_t)gsrc;
    v4u g0;
    g0[0] = 1u;                                               // count=1, user mode
    g0[1] = lds_off;                                          // LDS byte address
    g0[2] = (unsigned)ga;                                     // global addr [31:0]
    g0[3] = (unsigned)((ga >> 32) & 0x01FFFFFFu) | 0x80000000u; // addr[56:32] | type=2
    v8i g1;
    g1[0] = 0x00010000;            // workgroup_mask=0, data_size=1 (2 bytes)
    g1[1] = 0;                     // tensor_dim0[15:0]=0 (dim0 = 1<<30)
    g1[2] = 0x4000;                // tensor_dim0[31:16]=0x4000, tensor_dim1[15:0]=0
    g1[3] = (32 << 16) | 0x4000;   // tile_dim0=32 | tensor_dim1[31:16]=0x4000
    g1[4] = 64;                    // tile_dim1=64, tile_dim2=0
    g1[5] = 1024;                  // tensor_dim0_stride[31:0] = 1024 elements
    g1[6] = 0;                     // stride0 hi, stride1 lo
    g1[7] = 0;                     // stride1 hi
    v4i g2 = {0, 0, 0, 0};
    v4i g3 = {0, 0, 0, 0};
    v8i g4 = {0, 0, 0, 0, 0, 0, 0, 0};
    __builtin_amdgcn_tensor_load_to_lds(g0, g1, g2, g3, g4, 0);
}

// ---------------------------------------------------------------------------
// WMMA GEMM: G(4096 x 2048) = Xbf16(4096x1024) @ W(2048x1024)^T + (bih+bhh)
// Block = 8 waves computes 128(M) x 64(N). B panels staged in LDS by the
// Tensor Data Mover: 4 buffers, prefetch depth 2, ONE barrier per chunk
// (the buffer DMA'd at iter ch+2 was last read at iter ch-2, sequenced
// behind the top barrier of iter ch-1). A fragments direct from global.
// ---------------------------------------------------------------------------
__global__ __launch_bounds__(256)
void gemm_xw_wmma_kernel(const __bf16* __restrict__ X,
                         const __bf16* __restrict__ W,
                         const float* __restrict__ bih,
                         const float* __restrict__ bhh,
                         float* __restrict__ G) {
    __shared__ __align__(32) __bf16 ldsB[4][64][32];   // 4 x 4KB B panels

    const int lane = threadIdx.x & 31;
    const int wave = threadIdx.x >> 5;
    const int mb   = blockIdx.x >> 5;                  // 32 M-blocks of 128
    const int nbk  = blockIdx.x & 31;                  // 32 N-strips of 64
    const int m0 = mb * 128 + wave * 16;
    const int n0 = nbk * 64;

    const int hi  = lane >> 4;                         // 0: lanes 0-15, 1: 16-31
    const int l15 = lane & 15;

    const __bf16* wpanel = W + (size_t)n0 * DIM;       // rows n0..n0+63

    // prologue: DMA chunks 0,1 into buffers 0,1
    if (wave == 0) {
        tdm_load_panel(wpanel,      (unsigned)(uintptr_t)&ldsB[0][0][0]);
        tdm_load_panel(wpanel + 32, (unsigned)(uintptr_t)&ldsB[1][0][0]);
    }

    v8f acc0 = {}, acc1 = {}, acc2 = {}, acc3 = {};

    // A (16x32 bf16): lane L<16 holds M=L, K = {0..15}
    //                 lane L>=16 holds M=L-16, K = {16..31}
    const __bf16* xbase = X + (size_t)(m0 + l15) * DIM + (hi ? 16 : 0);

    for (int ch = 0; ch < KCHUNKS; ++ch) {
        const int k0 = ch * 32;
        if (wave == 0) {
            if (ch + 2 < KCHUNKS) {
                tdm_load_panel(wpanel + (size_t)(k0 + 64),
                               (unsigned)(uintptr_t)&ldsB[(ch + 2) & 3][0][0]);
                __builtin_amdgcn_s_wait_tensorcnt(2);  // chunk ch complete
            } else if (ch + 1 < KCHUNKS) {
                __builtin_amdgcn_s_wait_tensorcnt(1);  // chunk ch complete
            } else {
                __builtin_amdgcn_s_wait_tensorcnt(0);  // last chunk complete
            }
        }
        __syncthreads();   // publish panel ch; also guards buffer reuse

        v8bf alo = *(const v8bf*)(xbase + k0);
        v8bf ahi = *(const v8bf*)(xbase + k0 + 8);
        v16bf a = __builtin_shufflevector(alo, ahi,
                    0, 1, 2, 3, 4, 5, 6, 7, 8, 9, 10, 11, 12, 13, 14, 15);

        // B (32x16 bf16): lane holds column N = lane&15, K = j + (hi?16:0)
        const __bf16* bb = &ldsB[ch & 3][0][0];
        v16bf b0 = *(const v16bf*)(bb + (size_t)(l15)      * 32 + hi * 16);
        v16bf b1 = *(const v16bf*)(bb + (size_t)(16 + l15) * 32 + hi * 16);
        v16bf b2 = *(const v16bf*)(bb + (size_t)(32 + l15) * 32 + hi * 16);
        v16bf b3 = *(const v16bf*)(bb + (size_t)(48 + l15) * 32 + hi * 16);

        acc0 = __builtin_amdgcn_wmma_f32_16x16x32_bf16(false, a, false, b0, (short)0, acc0, false, false);
        acc1 = __builtin_amdgcn_wmma_f32_16x16x32_bf16(false, a, false, b1, (short)0, acc1, false, false);
        acc2 = __builtin_amdgcn_wmma_f32_16x16x32_bf16(false, a, false, b2, (short)0, acc2, false, false);
        acc3 = __builtin_amdgcn_wmma_f32_16x16x32_bf16(false, a, false, b3, (short)0, acc3, false, false);
    }

    // D layout: lane L, vgpr r -> row m0 + r + (hi?8:0), col n + (L&15)
    v8f accs[4] = {acc0, acc1, acc2, acc3};
#pragma unroll
    for (int nb = 0; nb < 4; ++nb) {
        const int n = n0 + nb * 16 + l15;
        const float bias = bih[n] + bhh[n];
#pragma unroll
        for (int r = 0; r < 8; ++r) {
            int m = m0 + r + (hi ? 8 : 0);
            G[(size_t)m * G4 + n] = accs[nb][r] + bias;
        }
    }
}

// ---------------------------------------------------------------------------
// Persistent BiLSTM recurrence.
// 32 workgroups: blockIdx/16 = direction, blockIdx%16 = shard.
// Each WG owns 32 hidden units (rows u, 512+u, 1024+u, 1536+u of Whh),
// keeps its 128x512 Whh shard packed as bf16 pairs in LDS,
// 128 threads: thread = gate*32 + unit. Global monotonic barrier per step.
// ---------------------------------------------------------------------------
__global__ __launch_bounds__(128)
void lstm_persist_kernel(const float* __restrict__ Gf, const float* __restrict__ Gb,
                         const float* __restrict__ Whf, const float* __restrict__ Whb,
                         const float* __restrict__ c0,
                         float* __restrict__ hf, float* __restrict__ hb,
                         float* __restrict__ hbuf, unsigned* __restrict__ bars) {
    const int dir = blockIdx.x >> 4;
    const int w   = blockIdx.x & 15;
    const int t   = threadIdx.x;          // 0..127
    const int uo  = t & 31;
    const int q   = t >> 5;               // gate: 0=i 1=f 2=g 3=o
    const int u   = w * 32 + uo;          // hidden unit 0..511
    const int row = q * H2 + u;           // 0..2047

    const float* G   = dir ? Gb  : Gf;
    const float* Whh = dir ? Whb : Whf;
    float* Hout      = dir ? hb  : hf;
    float* hdb       = hbuf + dir * 1024; // [2][512] double buffer
    unsigned* bar    = bars + dir;

    // 128 rows * 258 packed dwords (2 bf16 each, +2 pad dwords -> no bank conflicts)
    __shared__ unsigned lds_w[128 * 258];
    __shared__ float lds_h[H2];
    __shared__ float lds_z[128];

    // load + pack this thread's Whh row (f32 -> 2x bf16 per dword, truncate)
    {
        const float2* wr = (const float2*)(Whh + (size_t)row * H2);
        unsigned* dst = lds_w + t * 258;
        for (int k = 0; k < 256; ++k) {
            float2 v = wr[k];
            unsigned p = (__float_as_uint(v.y) & 0xffff0000u) |
                         (__float_as_uint(v.x) >> 16);
            dst[k] = p;
        }
    }
    float c = (t < 32) ? c0[dir * H2 + u] : 0.0f;
    __syncthreads();

    int cur = 0;
    for (int step = 0; step < S_LEN; ++step) {
        const int ts = dir ? (S_LEN - 1 - step) : step;

        // stage h into LDS
        for (int j = t; j < H2; j += 128) lds_h[j] = hdb[cur * H2 + j];
        __syncthreads();

        // z = G[ts][row] + dot(Whh[row], h)
        float acc = G[(size_t)ts * G4 + row];
        const unsigned* wrow = lds_w + t * 258;
#pragma unroll 8
        for (int k = 0; k < 256; ++k) {
            unsigned p = wrow[k];
            acc = fmaf(__uint_as_float(p << 16),         lds_h[2 * k],     acc);
            acc = fmaf(__uint_as_float(p & 0xffff0000u), lds_h[2 * k + 1], acc);
        }
        lds_z[t] = acc;
        __syncthreads();

        const int nxt = cur ^ 1;
        if (t < 32) {
            float zi = lds_z[t], zf = lds_z[32 + t], zg = lds_z[64 + t], zo = lds_z[96 + t];
            float gi = 1.0f / (1.0f + __expf(-zi));
            float gf = 1.0f / (1.0f + __expf(-zf));
            float gg = tanhf(zg);
            float go = 1.0f / (1.0f + __expf(-zo));
            c = gf * c + gi * gg;
            float hval = go * tanhf(c);
            const int uu = w * 32 + t;
            Hout[(size_t)ts * H2 + uu] = hval;
            hdb[nxt * H2 + uu] = hval;
        }
        // release + cross-WG barrier (monotonic counter)
        __builtin_amdgcn_fence(__ATOMIC_RELEASE, "agent");
        __syncthreads();
        if (t == 0) {
            __hip_atomic_fetch_add(bar, 1u, __ATOMIC_RELEASE, __HIP_MEMORY_SCOPE_AGENT);
            const unsigned target = (unsigned)NWG_DIR * (unsigned)(step + 1);
            while (__hip_atomic_load(bar, __ATOMIC_ACQUIRE, __HIP_MEMORY_SCOPE_AGENT) < target)
                __builtin_amdgcn_s_sleep(2);
        }
        __syncthreads();
        __builtin_amdgcn_fence(__ATOMIC_ACQUIRE, "agent");
        cur = nxt;
    }
}

// ---------------------------------------------------------------------------
// feats = [hf, hb] @ W_out.T + b_out   (N = 12, tiny -> VALU)
// ---------------------------------------------------------------------------
__global__ __launch_bounds__(32)
void out_proj_kernel(const float* __restrict__ hf, const float* __restrict__ hb,
                     const float* __restrict__ Wout, const float* __restrict__ bout,
                     float* __restrict__ feats) {
    const int tm = blockIdx.x;
    const int r  = threadIdx.x;
    if (r < T_TAGS) {
        const float* wr  = Wout + (size_t)r * 1024;
        const float* hfr = hf + (size_t)tm * H2;
        const float* hbr = hb + (size_t)tm * H2;
        float acc = bout[r];
        for (int k = 0; k < H2; ++k) acc = fmaf(hfr[k], wr[k], acc);
        for (int k = 0; k < H2; ++k) acc = fmaf(hbr[k], wr[H2 + k], acc);
        feats[tm * T_TAGS + r] = acc;
    }
}

// ---------------------------------------------------------------------------
// Viterbi max-plus DP + backtrack, single wave
// ---------------------------------------------------------------------------
__global__ __launch_bounds__(32)
void viterbi_kernel(const float* __restrict__ feats, const float* __restrict__ trans,
                    int* __restrict__ bptr, float* __restrict__ out) {
    __shared__ float fv[T_TAGS], nfv[T_TAGS], tr[T_TAGS * T_TAGS], term[T_TAGS];
    const int t = threadIdx.x;
    for (int i = t; i < T_TAGS * T_TAGS; i += 32) tr[i] = trans[i];
    if (t < T_TAGS) fv[t] = (t == START_T) ? 0.0f : NEGV;
    __syncthreads();
    for (int s = 0; s < S_LEN; ++s) {
        if (t < T_TAGS) {
            float best = -3.4e38f; int bi = 0;
            for (int p = 0; p < T_TAGS; ++p) {
                float sc = fv[p] + tr[t * T_TAGS + p];
                if (sc > best) { best = sc; bi = p; }   // first max, like argmax
            }
            nfv[t] = best + feats[s * T_TAGS + t];
            bptr[s * T_TAGS + t] = bi;
        }
        __syncthreads();
        if (t < T_TAGS) fv[t] = nfv[t];
        __syncthreads();
    }
    if (t < T_TAGS) term[t] = fv[t] + tr[STOP_T * T_TAGS + t];
    __syncthreads();
    if (t == 0) {
        float best = term[0]; int bi = 0;
        for (int p = 1; p < T_TAGS; ++p)
            if (term[p] > best) { best = term[p]; bi = p; }
        out[0] = best;                       // path_score
        int tag = bi;
        out[S_LEN] = (float)bi;              // best_path[S-1]
        for (int s = S_LEN - 1; s >= 1; --s) {
            tag = bptr[s * T_TAGS + tag];
            out[s] = (float)tag;             // best_path[s-1]
        }
    }
}

// ---------------------------------------------------------------------------
// launch
// ---------------------------------------------------------------------------
extern "C" void kernel_launch(void* const* d_in, const int* in_sizes, int n_in,
                              void* d_out, int out_size, void* d_ws, size_t ws_size,
                              hipStream_t stream) {
    const int*   sentence = (const int*)  d_in[0];
    const float* embed    = (const float*)d_in[1];
    const float* Wih_f    = (const float*)d_in[2];
    const float* Whh_f    = (const float*)d_in[3];
    const float* bih_f    = (const float*)d_in[4];
    const float* bhh_f    = (const float*)d_in[5];
    const float* Wih_b    = (const float*)d_in[6];
    const float* Whh_b    = (const float*)d_in[7];
    const float* bih_b    = (const float*)d_in[8];
    const float* bhh_b    = (const float*)d_in[9];
    const float* h0       = (const float*)d_in[10];
    const float* c0       = (const float*)d_in[11];
    const float* W_out    = (const float*)d_in[12];
    const float* b_out    = (const float*)d_in[13];
    const float* trans    = (const float*)d_in[14];
    float* out = (float*)d_out;

    // workspace carve-up (all offsets 256B aligned)
    char* ws = (char*)d_ws;
    size_t off = 0;
    __bf16* x_bf  = (__bf16*)(ws + off); off += (size_t)S_LEN * DIM * 2;        // 8 MB
    __bf16* Wf_bf = (__bf16*)(ws + off); off += (size_t)G4 * DIM * 2;           // 4 MB
    __bf16* Wb_bf = (__bf16*)(ws + off); off += (size_t)G4 * DIM * 2;           // 4 MB
    float*  Gf    = (float*) (ws + off); off += (size_t)S_LEN * G4 * 4;         // 32 MB
    float*  Gb    = (float*) (ws + off); off += (size_t)S_LEN * G4 * 4;         // 32 MB
    float*  hf    = (float*) (ws + off); off += (size_t)S_LEN * H2 * 4;         // 8 MB
    float*  hb    = (float*) (ws + off); off += (size_t)S_LEN * H2 * 4;         // 8 MB
    float*  feats = (float*) (ws + off); off += (size_t)S_LEN * T_TAGS * 4;
    int*    bptr  = (int*)   (ws + off); off += (size_t)S_LEN * T_TAGS * 4;
    float*  hbuf  = (float*) (ws + off); off += 2 * 2 * H2 * 4;
    unsigned* bars= (unsigned*)(ws + off); off += 256;

    // 1) init barrier counters + h double buffers
    init_state_kernel<<<1, 1024, 0, stream>>>(h0, hbuf, bars);

    // 2) embedding gather -> bf16
    gather_embed_kernel<<<S_LEN, 1024, 0, stream>>>(sentence, embed, x_bf);

    // 3) weight conversion -> bf16
    {
        const int n = G4 * DIM;
        cvt_bf16_kernel<<<(n + 255) / 256, 256, 0, stream>>>(Wih_f, Wf_bf, n);
        cvt_bf16_kernel<<<(n + 255) / 256, 256, 0, stream>>>(Wih_b, Wb_bf, n);
    }

    // 4) WMMA input-projection GEMMs (G = x @ Wih^T + bih + bhh), TDM-staged B
    {
        const int blocks = (S_LEN / 128) * (G4 / 64);   // 32 * 32 = 1024
        gemm_xw_wmma_kernel<<<blocks, 256, 0, stream>>>(x_bf, Wf_bf, bih_f, bhh_f, Gf);
        gemm_xw_wmma_kernel<<<blocks, 256, 0, stream>>>(x_bf, Wb_bf, bih_b, bhh_b, Gb);
    }

    // 5) persistent bidirectional recurrence (both directions concurrently)
    lstm_persist_kernel<<<2 * NWG_DIR, 128, 0, stream>>>(Gf, Gb, Whh_f, Whh_b, c0,
                                                         hf, hb, hbuf, bars);

    // 6) output projection
    out_proj_kernel<<<S_LEN, 32, 0, stream>>>(hf, hb, W_out, b_out, feats);

    // 7) Viterbi DP + backtrack
    viterbi_kernel<<<1, 32, 0, stream>>>(feats, trans, bptr, out);

    (void)in_sizes; (void)n_in; (void)out_size; (void)ws_size;
}
